// EfficientChannelGroupAttention_4466765987898
// MI455X (gfx1250) — matrix operand order repaired and checked by reading, not compiled, not run on hardware
//
#include <hip/hip_runtime.h>

// ---------------------------------------------------------------------------
// EfficientChannelGroupAttention for MI455X (gfx1250, wave32, WMMA + TDM)
//   B=16, N=1024, C=768, G=4, H=12, M=64, Cg=192, CC=384, cpg=96, hd=8
// Pipeline:
//   1) convert weights f32->bf16 (once, into ws)
//   2) q/k/v grouped projections  (WMMA bf16, f32 acc)  -> bf16 [BG,N,96]
//   3) bank projections kb/vb     (WMMA bf16)           -> bf16 [B,M,96]
//   4) flash attention, 8 query tiles per block; K/V head-slices staged to
//      LDS once per block via TDM tensor_load_to_lds (TENSORcnt); Q tile via
//      global_load_async_to_lds_b64 (ASYNCcnt); WMMA for scores and P@V
//   5) output projection          (WMMA bf16) + bias    -> f32 d_out [B,N,768]
// ---------------------------------------------------------------------------

typedef __bf16 bf16;
typedef bf16  v16bf __attribute__((ext_vector_type(16)));
typedef float v8f   __attribute__((ext_vector_type(8)));
typedef unsigned int u32x4 __attribute__((ext_vector_type(4)));
typedef int i32x4 __attribute__((ext_vector_type(4)));
typedef int i32x8 __attribute__((ext_vector_type(8)));

constexpr int B_  = 16;
constexpr int N_  = 1024;
constexpr int C_  = 768;
constexpr int G_  = 4;
constexpr int H_  = 12;
constexpr int M_  = 64;
constexpr int Cg_  = C_ / G_;    // 192
constexpr int CC_  = C_ / 2;     // 384
constexpr int cpg_ = CC_ / G_;   // 96
constexpr int hd_  = cpg_ / H_;  // 8
constexpr int BG_  = B_ * G_;    // 64
constexpr int NKEY_ = N_ + M_;   // 1088
constexpr int QT_  = 8;          // query tiles (of 16 rows) per attention block

#ifdef __has_builtin
#  if __has_builtin(__builtin_amdgcn_tensor_load_to_lds)
#    define HAVE_TDM 1
#  endif
#  if __has_builtin(__builtin_amdgcn_s_wait_tensorcnt)
#    define HAVE_WAIT_TENSOR 1
#  endif
#endif
#ifndef HAVE_TDM
#  define HAVE_TDM 0
#endif
#ifndef HAVE_WAIT_TENSOR
#  define HAVE_WAIT_TENSOR 0
#endif

__device__ __forceinline__ bf16 f2bf(float f) {
    unsigned u = __builtin_bit_cast(unsigned, f);
    u += 0x7fffu + ((u >> 16) & 1u);            // round-to-nearest-even
    return __builtin_bit_cast(bf16, (unsigned short)(u >> 16));
}
__device__ __forceinline__ bf16 bzero() {
    return __builtin_bit_cast(bf16, (unsigned short)0);
}

#if HAVE_TDM
// 2-D TDM tile load: copies tile_d1 rows of tile_d0 bf16 elements (row stride
// stride0 elements in global memory) into contiguous LDS at lds_addr.
// Descriptor packing per CDNA5 ISA:
//   group0: count=1, lds_addr[63:32], global_addr[120:64], type=2 [127:126]
//   group1: data_size=1 (2B) [17:16], tensor_dim0 [79:48], tensor_dim1
//           [111:80], tile_dim0 [127:112], tile_dim1 [143:128],
//           tensor_dim0_stride [207:160]
__device__ __forceinline__ void tdm_load_2d(unsigned lds_addr, const void* gptr,
                                            unsigned tensor_d0, unsigned tensor_d1,
                                            unsigned long long stride0,
                                            unsigned tile_d0, unsigned tile_d1)
{
    const unsigned long long ga = (unsigned long long)gptr;
    u32x4 g0;
    g0[0] = 1u;                                               // count=1
    g0[1] = lds_addr;                                         // lds_addr
    g0[2] = (unsigned)(ga & 0xffffffffu);                     // global_addr lo
    g0[3] = (unsigned)((ga >> 32) & 0x01ffffffu) | 0x80000000u; // ga hi | type=2
    i32x8 g1;
    g1[0] = (int)(1u << 16);                                  // data_size=2B
    g1[1] = (int)((tensor_d0 & 0xffffu) << 16);               // td0[15:0]
    g1[2] = (int)(((tensor_d0 >> 16) & 0xffffu) | ((tensor_d1 & 0xffffu) << 16));
    g1[3] = (int)(((tensor_d1 >> 16) & 0xffffu) | ((tile_d0 & 0xffffu) << 16));
    g1[4] = (int)(tile_d1 & 0xffffu);                         // tile_dim2=0
    g1[5] = (int)(stride0 & 0xffffffffu);
    g1[6] = (int)((stride0 >> 32) & 0xffffu);                 // stride1=0
    g1[7] = 0;
    const i32x4 gz = {0, 0, 0, 0};
#if __clang_major__ >= 23
    const i32x8 gz8 = {0, 0, 0, 0, 0, 0, 0, 0};
    __builtin_amdgcn_tensor_load_to_lds(g0, g1, gz, gz, gz8, 0);
#else
    __builtin_amdgcn_tensor_load_to_lds(g0, g1, gz, gz, 0);
#endif
}
#endif

__device__ __forceinline__ void wait_tensorcnt0() {
#if HAVE_WAIT_TENSOR
    __builtin_amdgcn_s_wait_tensorcnt((short)0);
#else
    asm volatile("s_wait_tensorcnt 0x0" ::: "memory");
#endif
}

// ---------------------------------------------------------------------------
// elementwise f32 -> bf16 conversion (weights)
// ---------------------------------------------------------------------------
__global__ void cvt_f32_bf16(const float* __restrict__ src, bf16* __restrict__ dst, int n) {
    int i = blockIdx.x * 256 + threadIdx.x;
    if (i < n) dst[i] = f2bf(src[i]);
}

// ---------------------------------------------------------------------------
// Grouped projection: Y[r, nc] = sum_k Xrow(r)[k] * W[nc][k] + bias[nc]
// One wave per 16x16 output tile, WMMA 16x16x32 bf16, f32 X input.
// ---------------------------------------------------------------------------
__global__ __launch_bounds__(32)
void proj_f32_bf16(const float* __restrict__ X, const bf16* __restrict__ W,
                   const float* __restrict__ bias, bf16* __restrict__ Y,
                   int rows_per_seg, int Cfull, int Kdim, int Gdiv, int Nout)
{
    const int lane = threadIdx.x;
    const int half = lane >> 4;
    const int l16  = lane & 15;
    const int r0   = blockIdx.x * 16;
    const int r    = r0 + l16;                    // A-frag row (M = l16)
    const int seg  = r / rows_per_seg;
    const int nin  = r % rows_per_seg;
    const int bidx = seg / Gdiv;
    const int g    = seg % Gdiv;
    const float* xrow = X + ((size_t)(bidx * rows_per_seg + nin)) * Cfull + (size_t)g * Kdim;
    const int   ncol  = blockIdx.y * 16 + l16;    // B-frag / D column
    const bf16* wrow  = W + (size_t)ncol * Kdim;

    v8f acc = {};
    for (int kk = 0; kk < Kdim; kk += 32) {
        if (kk + 32 < Kdim) {
            __builtin_prefetch(xrow + kk + 32, 0, 0);
            __builtin_prefetch(wrow + kk + 32, 0, 0);
        }
        v16bf a, b;
#pragma unroll
        for (int i = 0; i < 8; ++i) {
            const int ka = kk + ((i >= 4) ? 16 : 0) + half * 8 + (i & 3) * 2;
            a[2*i]   = f2bf(xrow[ka]);
            a[2*i+1] = f2bf(xrow[ka + 1]);
            const int kb = kk + ((i >= 4) ? 16 : 0) + half * 8 + (i & 3) * 2;
            b[2*i]   = wrow[kb];
            b[2*i+1] = wrow[kb + 1];
        }
        acc = __builtin_amdgcn_wmma_f32_16x16x32_bf16(false, a, false, b,
                                                      (short)0, acc, false, false);
    }
    const float bv = bias[ncol];
#pragma unroll
    for (int rr = 0; rr < 8; ++rr) {
        const int mrow = rr + 8 * half;
        Y[(size_t)(r0 + mrow) * Nout + ncol] = f2bf(acc[rr] + bv);
    }
}

// ---------------------------------------------------------------------------
// Flat projection, bf16 A input, f32 output with bias (final projection).
// ---------------------------------------------------------------------------
__global__ __launch_bounds__(32)
void proj_bf16_f32(const bf16* __restrict__ X, const bf16* __restrict__ W,
                   const float* __restrict__ bias, float* __restrict__ Y,
                   int Kdim, int Nout)
{
    const int lane = threadIdx.x;
    const int half = lane >> 4;
    const int l16  = lane & 15;
    const int r0   = blockIdx.x * 16;
    const bf16* xrow = X + (size_t)(r0 + l16) * Kdim;
    const int   ncol = blockIdx.y * 16 + l16;
    const bf16* wrow = W + (size_t)ncol * Kdim;

    v8f acc = {};
    for (int kk = 0; kk < Kdim; kk += 32) {
        if (kk + 32 < Kdim) {
            __builtin_prefetch(xrow + kk + 32, 0, 0);
            __builtin_prefetch(wrow + kk + 32, 0, 0);
        }
        v16bf a, b;
#pragma unroll
        for (int i = 0; i < 8; ++i) {
            const int ka = kk + ((i >= 4) ? 16 : 0) + half * 8 + (i & 3) * 2;
            a[2*i]   = xrow[ka];
            a[2*i+1] = xrow[ka + 1];
            const int kb = kk + ((i >= 4) ? 16 : 0) + half * 8 + (i & 3) * 2;
            b[2*i]   = wrow[kb];
            b[2*i+1] = wrow[kb + 1];
        }
        acc = __builtin_amdgcn_wmma_f32_16x16x32_bf16(false, a, false, b,
                                                      (short)0, acc, false, false);
    }
    const float bv = bias[ncol];
#pragma unroll
    for (int rr = 0; rr < 8; ++rr) {
        const int mrow = rr + 8 * half;
        Y[(size_t)(r0 + mrow) * Nout + ncol] = acc[rr] + bv;
    }
}

// ---------------------------------------------------------------------------
// Flash attention: grid = (N/(16*QT), H, BG); one wave per block.
// K/V head-slices [1088 x 8] bf16 staged into LDS once per block with TDM
// (4 descriptors: main + bank for each of K and V), then 8 query tiles are
// processed against LDS-resident keys. Q tile staged per-qtile with an async
// global->LDS copy. hd=8 is zero-padded inside the K=32 WMMA fragments.
// ---------------------------------------------------------------------------
__global__ __launch_bounds__(32)
void attn_kernel(const bf16* __restrict__ Q, const bf16* __restrict__ Kq,
                 const bf16* __restrict__ V, const bf16* __restrict__ KB,
                 const bf16* __restrict__ VB, bf16* __restrict__ O)
{
    __shared__ __align__(16) bf16 klds[NKEY_ * hd_];   // [key][d]  17408 B
    __shared__ __align__(16) bf16 vlds[NKEY_ * hd_];   // [key][d]  17408 B
    __shared__ __align__(16) bf16 qlds[16 * hd_];      // [row][d]    256 B
    __shared__ __align__(16) bf16 pbuf[16 * 32];       // [row][key] 1024 B

    const int lane = threadIdx.x;
    const int half = lane >> 4;
    const int l16  = lane & 15;
    const int h  = blockIdx.y;
    const int bg = blockIdx.z;
    const int b  = bg / G_;
    const int g  = bg % G_;
    const float scale = 0.35355339059327373f;   // 1/sqrt(hd)
    const bf16 bz = bzero();

    const bf16* ksrc_main = Kq + ((size_t)bg * N_) * cpg_ + h * hd_;
    const bf16* ksrc_bank = KB + ((size_t)b * M_) * cpg_ + h * hd_;
    const bf16* vsrc_main = V  + ((size_t)bg * N_) * cpg_ + h * hd_;
    const bf16* vsrc_bank = VB + ((size_t)b * M_) * cpg_ + h * hd_;

#if HAVE_TDM
    // Tensor Data Mover: 2-D strided tiles -> contiguous LDS (TENSORcnt).
    tdm_load_2d((unsigned)(size_t)klds,              ksrc_main, hd_, N_, cpg_, hd_, N_);
    tdm_load_2d((unsigned)(size_t)(klds + N_ * hd_), ksrc_bank, hd_, M_, cpg_, hd_, M_);
    tdm_load_2d((unsigned)(size_t)vlds,              vsrc_main, hd_, N_, cpg_, hd_, N_);
    tdm_load_2d((unsigned)(size_t)(vlds + N_ * hd_), vsrc_bank, hd_, M_, cpg_, hd_, M_);
    wait_tensorcnt0();
#else
    for (int r = lane; r < NKEY_; r += 32) {
        const bf16* kr = (r < N_) ? (ksrc_main + (size_t)r * cpg_)
                                  : (ksrc_bank + (size_t)(r - N_) * cpg_);
        const bf16* vr = (r < N_) ? (vsrc_main + (size_t)r * cpg_)
                                  : (vsrc_bank + (size_t)(r - N_) * cpg_);
#pragma unroll
        for (int d = 0; d < 8; ++d) { klds[r * 8 + d] = kr[d]; vlds[r * 8 + d] = vr[d]; }
    }
#endif
    __syncthreads();

    for (int qi = 0; qi < QT_; ++qi) {
        const int n0 = (blockIdx.x * QT_ + qi) * 16;

        // ---- async global->LDS copy of the 16x8 Q tile (ASYNCcnt) ----
        {
            const bf16* qsrc = Q + ((size_t)(bg * N_ + n0 + (lane >> 1))) * cpg_
                                 + h * hd_ + (lane & 1) * 4;
            const unsigned dst = (unsigned)(size_t)qlds + (unsigned)lane * 8u;
            asm volatile("global_load_async_to_lds_b64 %0, %1, off"
                         :: "v"(dst), "v"((unsigned long long)(size_t)qsrc)
                         : "memory");
            asm volatile("s_wait_asynccnt 0x0" ::: "memory");
        }
        __syncthreads();

        // ---- Q fragment (dims 0..7 valid, rest zero) ----
        v16bf qa;
#pragma unroll
        for (int i = 0; i < 16; ++i) qa[i] = bz;
        if (half == 0) {
#pragma unroll
            for (int i = 0; i < 4; ++i) {
                qa[2*i]   = qlds[l16 * 8 + 2*i];
                qa[2*i+1] = qlds[l16 * 8 + 2*i + 1];
            }
        }

        float rmax[8], rsum[8];
#pragma unroll
        for (int rr = 0; rr < 8; ++rr) { rmax[rr] = -3.0e38f; rsum[rr] = 0.0f; }
        v8f co = {};

        for (int j0 = 0; j0 < NKEY_; j0 += 32) {
            // ---- scores: two 16-key B fragments from LDS ----
            v16bf kf0, kf1;
#pragma unroll
            for (int i = 0; i < 16; ++i) { kf0[i] = bz; kf1[i] = bz; }
            if (half == 0) {
                const int ja = (j0 + l16) * 8;
                const int jb = (j0 + 16 + l16) * 8;
#pragma unroll
                for (int i = 0; i < 4; ++i) {
                    kf0[2*i] = klds[ja + 2*i]; kf0[2*i+1] = klds[ja + 2*i + 1];
                    kf1[2*i] = klds[jb + 2*i]; kf1[2*i+1] = klds[jb + 2*i + 1];
                }
            }
            v8f s0 = {}, s1 = {};
            s0 = __builtin_amdgcn_wmma_f32_16x16x32_bf16(false, qa, false, kf0,
                                                         (short)0, s0, false, false);
            s1 = __builtin_amdgcn_wmma_f32_16x16x32_bf16(false, qa, false, kf1,
                                                         (short)0, s1, false, false);

            // ---- online softmax update, per owned row (rr + 8*half) ----
#pragma unroll
            for (int rr = 0; rr < 8; ++rr) {
                float x0 = s0[rr] * scale;
                float x1 = s1[rr] * scale;
                float tm = fmaxf(x0, x1);
#pragma unroll
                for (int off = 1; off < 16; off <<= 1)
                    tm = fmaxf(tm, __shfl_xor(tm, off, 32));
                const float nm  = fmaxf(rmax[rr], tm);
                const float fac = __expf(rmax[rr] - nm);
                const float p0  = __expf(x0 - nm);
                const float p1  = __expf(x1 - nm);
                float ts = p0 + p1;
#pragma unroll
                for (int off = 1; off < 16; off <<= 1)
                    ts += __shfl_xor(ts, off, 32);
                rsum[rr] = rsum[rr] * fac + ts;
                rmax[rr] = nm;
                co[rr] *= fac;
                const int mrow = rr + 8 * half;
                pbuf[mrow * 32 + l16]      = f2bf(p0);
                pbuf[mrow * 32 + 16 + l16] = f2bf(p1);
            }
            __syncthreads();

            // ---- P @ V: A = probs [16 x 32 keys], B = V [32 keys x 16(hd=8)]
            v16bf pa, vb;
#pragma unroll
            for (int i = 0; i < 8; ++i) {
                const int ka = ((i >= 4) ? 16 : 0) + half * 8 + (i & 3) * 2;
                pa[2*i]   = pbuf[l16 * 32 + ka];
                pa[2*i+1] = pbuf[l16 * 32 + ka + 1];
                const int key = j0 + ((i >= 4) ? 16 : 0) + half * 8 + (i & 3) * 2;
                if (l16 < 8) {
                    vb[2*i]   = vlds[key * 8 + l16];
                    vb[2*i+1] = vlds[(key + 1) * 8 + l16];
                } else {
                    vb[2*i] = bz; vb[2*i+1] = bz;
                }
            }
            co = __builtin_amdgcn_wmma_f32_16x16x32_bf16(false, pa, false, vb,
                                                         (short)0, co, false, false);
        }

        // ---- normalize and write merged output: O[b][n][g*96 + h*8 + d] ----
        if (l16 < 8) {
#pragma unroll
            for (int rr = 0; rr < 8; ++rr) {
                const int mrow = rr + 8 * half;
                const float val = co[rr] / rsum[rr];
                O[((size_t)(b * N_ + n0 + mrow)) * CC_ + g * cpg_ + h * hd_ + l16]
                    = f2bf(val);
            }
        }
        __syncthreads();
    }
}

// ---------------------------------------------------------------------------
extern "C" void kernel_launch(void* const* d_in, const int* in_sizes, int n_in,
                              void* d_out, int out_size, void* d_ws, size_t ws_size,
                              hipStream_t stream) {
    const float* x      = (const float*)d_in[0];
    const float* k_bank = (const float*)d_in[1];
    const float* v_bank = (const float*)d_in[2];
    const float* q_w  = (const float*)d_in[3];  const float* q_b  = (const float*)d_in[4];
    const float* k_w  = (const float*)d_in[5];  const float* k_b  = (const float*)d_in[6];
    const float* v_w  = (const float*)d_in[7];  const float* v_b  = (const float*)d_in[8];
    const float* bk_w = (const float*)d_in[9];  const float* bk_b = (const float*)d_in[10];
    const float* bv_w = (const float*)d_in[11]; const float* bv_b = (const float*)d_in[12];
    const float* p_w  = (const float*)d_in[13]; const float* p_b  = (const float*)d_in[14];
    float* out = (float*)d_out;

    // ---- workspace carving (bf16, 256B aligned chunks) ----
    char* base = (char*)d_ws;
    size_t off = 0;
    auto take = [&](size_t elems) -> bf16* {
        bf16* r = (bf16*)(base + off);
        off += ((elems * sizeof(bf16) + 255) & ~(size_t)255);
        return r;
    };
    bf16* qw  = take((size_t)cpg_ * Cg_);
    bf16* kw  = take((size_t)cpg_ * Cg_);
    bf16* vw  = take((size_t)cpg_ * Cg_);
    bf16* bkw = take((size_t)cpg_ * C_);
    bf16* bvw = take((size_t)cpg_ * C_);
    bf16* pw  = take((size_t)C_ * CC_);
    bf16* q_all = take((size_t)BG_ * N_ * cpg_);
    bf16* k_all = take((size_t)BG_ * N_ * cpg_);
    bf16* v_all = take((size_t)BG_ * N_ * cpg_);
    bf16* kbp   = take((size_t)B_ * M_ * cpg_);
    bf16* vbp   = take((size_t)B_ * M_ * cpg_);
    bf16* om    = take((size_t)B_ * N_ * CC_);

    // ---- 1) weight conversion ----
    auto cvt = [&](const float* s, bf16* d, int n) {
        cvt_f32_bf16<<<dim3((n + 255) / 256), dim3(256), 0, stream>>>(s, d, n);
    };
    cvt(q_w,  qw,  cpg_ * Cg_);
    cvt(k_w,  kw,  cpg_ * Cg_);
    cvt(v_w,  vw,  cpg_ * Cg_);
    cvt(bk_w, bkw, cpg_ * C_);
    cvt(bv_w, bvw, cpg_ * C_);
    cvt(p_w,  pw,  C_ * CC_);

    const dim3 wave(32);

    // ---- 2) q/k/v grouped projections: rows = BG*N, K = Cg ----
    {
        dim3 grid((BG_ * N_) / 16, cpg_ / 16);
        proj_f32_bf16<<<grid, wave, 0, stream>>>(x, qw, q_b, q_all, N_, C_, Cg_, G_, cpg_);
        proj_f32_bf16<<<grid, wave, 0, stream>>>(x, kw, k_b, k_all, N_, C_, Cg_, G_, cpg_);
        proj_f32_bf16<<<grid, wave, 0, stream>>>(x, vw, v_b, v_all, N_, C_, Cg_, G_, cpg_);
    }
    // ---- 3) bank projections: rows = B*M, K = C ----
    {
        dim3 grid((B_ * M_) / 16, cpg_ / 16);
        proj_f32_bf16<<<grid, wave, 0, stream>>>(k_bank, bkw, bk_b, kbp, M_, C_, C_, 1, cpg_);
        proj_f32_bf16<<<grid, wave, 0, stream>>>(v_bank, bvw, bv_b, vbp, M_, C_, C_, 1, cpg_);
    }
    // ---- 4) attention: 8 query tiles per block, K/V staged via TDM ----
    {
        dim3 grid(N_ / (16 * QT_), H_, BG_);
        attn_kernel<<<grid, wave, 0, stream>>>(q_all, k_all, v_all, kbp, vbp, om);
    }
    // ---- 5) final projection: rows = B*N, K = CC, Nout = C ----
    {
        dim3 grid((B_ * N_) / 16, C_ / 16);
        proj_bf16_f32<<<grid, wave, 0, stream>>>(om, pw, p_b, out, CC_, C_);
    }
    (void)in_sizes; (void)n_in; (void)out_size; (void)ws_size;
}
